// VQVAE_4157528343203
// MI455X (gfx1250) — matrix operand order repaired and checked by reading, not compiled, not run on hardware
//
#include <hip/hip_runtime.h>
#include <hip/hip_bf16.h>

// ---------------- problem constants ----------------
#define N_NODES   60000
#define N_EDGES   240000
#define EMB       300
#define HID       600
#define NUM_LAYERS 5
#define NUM_TOKENS 512
#define BN_EPS    1e-5f
// padded sizes
#define MP    60032      // rows: 938 * 64
#define EMBP  320        // K pad of 300 (mult 32)
#define HIDP  608        // K pad of 600 (mult 32)
#define YW    640        // Y stored width (5 * 128, unguarded GEMM1 stores)
#define NT1   640        // Bt rows for GEMM1 (ceil(608/128)*128)
#define NT2   384        // Bt rows for GEMM2 (ceil(320/128)*128)

typedef __bf16 bf16_t;
typedef __attribute__((ext_vector_type(16))) __bf16 v16bf;
typedef __attribute__((ext_vector_type(8)))  float  v8f;

union Frag16 { uint4 q[2]; v16bf v; };        // one lane's 16 bf16 (32 B)
union CFrag  { v8f v; float e[8]; };

// ---------------- small helper kernels ----------------
__global__ void k_zero_buf(float* __restrict__ p, int n) {
    int i = blockIdx.x * blockDim.x + threadIdx.x;
    if (i < n) p[i] = 0.f;
}

// Build B^T panel: dst[n*Kp + k] = B[k][n] (bf16, zero padded).
// src_kn==1: src is row-major (K x N)  -> src[k*N+n]   (weights)
// src_kn==0: src is row-major (N x K)  -> src[n*K+k]   (codebook: B = cb^T so B^T = cb)
__global__ void k_pack_bt(const float* __restrict__ src, bf16_t* __restrict__ dst,
                          int K, int N, int Kp, int Nalloc, int src_kn) {
    int idx = blockIdx.x * blockDim.x + threadIdx.x;
    if (idx >= Nalloc * Kp) return;
    int n = idx / Kp, k = idx % Kp;
    float v = 0.f;
    if (k < K && n < N) v = src_kn ? src[k * N + n] : src[n * K + k];
    dst[idx] = (bf16_t)v;
}

// h0[i][c] = x_emb1[atom_type][c] + x_emb2[chirality][c]; zero in padded region
__global__ void k_init_h(const int* __restrict__ x,
                         const float* __restrict__ e1, const float* __restrict__ e2,
                         float* __restrict__ h) {
    int idx = blockIdx.x * blockDim.x + threadIdx.x;
    if (idx >= MP * EMBP) return;
    int i = idx / EMBP, c = idx % EMBP;
    float v = 0.f;
    if (i < N_NODES && c < EMB) {
        int at = x[2 * i], ch = x[2 * i + 1];
        v = e1[at * EMB + c] + e2[ch * EMB + c];
    }
    h[idx] = v;
}

// aggr = h + self_ee (self-loop message, atomic-free); zero padded region
__global__ void k_aggr_init(const float* __restrict__ h,
                            const float* __restrict__ ee1, const float* __restrict__ ee2,
                            float* __restrict__ aggr) {
    int idx = blockIdx.x * blockDim.x + threadIdx.x;
    if (idx >= MP * EMBP) return;
    int i = idx / EMBP, c = idx % EMBP;
    float v = 0.f;
    if (i < N_NODES && c < EMB)
        v = h[idx] + ee1[4 * EMB + c] + ee2[c];
    aggr[idx] = v;
}

// one wave (32 lanes) per edge: aggr[dst] += h[src] + ee1[bt] + ee2[bd]
__global__ __launch_bounds__(256)
void k_scatter(const float* __restrict__ h, const int* __restrict__ ei,
               const int* __restrict__ ea,
               const float* __restrict__ ee1, const float* __restrict__ ee2,
               float* __restrict__ aggr) {
    int e = blockIdx.x * 8 + (threadIdx.x >> 5);
    if (e >= N_EDGES) return;
    int lane = threadIdx.x & 31;
    int s  = ei[e];
    int d  = ei[N_EDGES + e];
    int bt = ea[2 * e];
    int bd = ea[2 * e + 1];
    const float* hs = h + (size_t)s * EMBP;
    float*       ad = aggr + (size_t)d * EMBP;
    __builtin_prefetch(hs, 0, 1);              // gfx1250 global_prefetch_b8
    for (int c = lane; c < EMB; c += 32)
        atomicAdd(&ad[c], hs[c] + ee1[bt * EMB + c] + ee2[bd * EMB + c]);
}

// ---------------- WMMA bf16 GEMM ----------------
// C(MP x Nc) = A(MP x Ktot) @ B(Ktot x Nc) [+bias] [relu]; B supplied transposed
// (Bt: Nalloc x Ktot, zero padded). A rows fully padded -> no load guards at all.
// Block: 256 thr = 8 waves in 2x4 -> 64x128 C tile; wave: 32x32, 4 wmma / K-step.
// LDS is held in *fragment order*: [tile][lane][16 bf16] so every fragment
// load is two ds_load_b128 and every staging store is one ds_store_b128.
template<bool A_BF16, bool HAS_BIAS, bool RELU, bool OUT_BF16>
__global__ __launch_bounds__(256, 2)
void k_gemm(const void* __restrict__ Aptr, int lda,
            const bf16_t* __restrict__ Bt,
            const float* __restrict__ bias, int nbias,
            void* __restrict__ Cptr, int ldc, int Ktot) {
    __shared__ uint4 AsF[4 * 32 * 2];   // 4 m-tiles x 32 lanes x 32B = 4 KB
    __shared__ uint4 BsF[8 * 32 * 2];   // 8 n-tiles x 32 lanes x 32B = 8 KB

    const int tid  = threadIdx.x;
    const int lane = tid & 31;
    const int wave = tid >> 5;
    const int wm = wave >> 2, wn = wave & 3;
    const int g = lane >> 4, lm = lane & 15;
    const int rowBase = blockIdx.x * 64;
    const int colBase = blockIdx.y * 128;

    // ---- staging maps ----
    // A: thread -> (row ar, 8 consecutive K at ac0). A-layout (matches B): lane group g
    // holds K = 16g + e in e0..15, so the 8-value run lands
    // contiguously in one half of one lane's fragment.
    const int ar  = tid >> 2;                 // 0..63
    const int ac0 = (tid & 3) * 8;            // 0,8,16,24
    const int a_dst = ((ar >> 4) * 32 + ((ar & 15) + ((ac0 & 16) ? 16 : 0))) * 2
                    + ((ac0 & 8) ? 1 : 0);
    // B: thread -> (column bc, K half bg). B-layout (contiguous k order, same as A): lane (n + 16g) holds
    // K = 16g + e, e contiguous -> one Bt row segment is one full fragment.
    const int bc  = tid >> 1;                 // 0..127
    const int bg  = tid & 1;                  // 0/1 -> K half
    const int b_dst = ((bc >> 4) * 32 + ((bc & 15) + bg * 16)) * 2;

    const float*  Af = (const float*)Aptr;
    const bf16_t* Ab = (const bf16_t*)Aptr;
    const size_t  aRowOff = (size_t)(rowBase + ar) * lda + ac0;
    const bf16_t* btRow   = Bt + (size_t)(colBase + bc) * Ktot + bg * 16;

    auto loadA = [&](int kk) -> uint4 {
        if (A_BF16) {
            return *(const uint4*)(Ab + aRowOff + kk);
        } else {
            const float4* p = (const float4*)(Af + aRowOff + kk);
            float4 f0 = p[0], f1 = p[1];
            union { bf16_t h[8]; uint4 u; } cv;
            cv.h[0] = (bf16_t)f0.x; cv.h[1] = (bf16_t)f0.y;
            cv.h[2] = (bf16_t)f0.z; cv.h[3] = (bf16_t)f0.w;
            cv.h[4] = (bf16_t)f1.x; cv.h[5] = (bf16_t)f1.y;
            cv.h[6] = (bf16_t)f1.z; cv.h[7] = (bf16_t)f1.w;
            return cv.u;
        }
    };

    CFrag acc[2][2];
#pragma unroll
    for (int a = 0; a < 2; ++a)
#pragma unroll
        for (int b = 0; b < 2; ++b)
#pragma unroll
            for (int r = 0; r < 8; ++r) acc[a][b].e[r] = 0.f;

    // software pipeline: global loads for step k+32 overlap the WMMAs of step k
    uint4 aCur = loadA(0);
    uint4 bCur0 = ((const uint4*)btRow)[0];
    uint4 bCur1 = ((const uint4*)btRow)[1];

    for (int k0 = 0; k0 < Ktot; k0 += 32) {
        AsF[a_dst]     = aCur;
        BsF[b_dst]     = bCur0;
        BsF[b_dst + 1] = bCur1;
        __syncthreads();

        int kn = k0 + 32;
        if (kn < Ktot) {
            aCur = loadA(kn);
            const uint4* bp = (const uint4*)(btRow + kn);
            bCur0 = bp[0]; bCur1 = bp[1];
        }

        Frag16 a[2], b[2];
#pragma unroll
        for (int f = 0; f < 2; ++f) {
            int ai = ((wm * 2 + f) * 32 + lane) * 2;
            a[f].q[0] = AsF[ai]; a[f].q[1] = AsF[ai + 1];
        }
#pragma unroll
        for (int f = 0; f < 2; ++f) {
            int bi = ((wn * 2 + f) * 32 + lane) * 2;
            b[f].q[0] = BsF[bi]; b[f].q[1] = BsF[bi + 1];
        }
#pragma unroll
        for (int fm = 0; fm < 2; ++fm)
#pragma unroll
            for (int fn = 0; fn < 2; ++fn)
                acc[fm][fn].v = __builtin_amdgcn_wmma_f32_16x16x32_bf16(
                    false, a[fm].v, false, b[fn].v,
                    (short)0, acc[fm][fn].v, false, false);
        __syncthreads();
    }

    // epilogue: C layout VGPR r -> M = r + 8*(lane/16), N = lane%16
#pragma unroll
    for (int fm = 0; fm < 2; ++fm)
#pragma unroll
        for (int fn = 0; fn < 2; ++fn)
#pragma unroll
            for (int r = 0; r < 8; ++r) {
                int row = rowBase + wm * 32 + fm * 16 + g * 8 + r;
                int col = colBase + wn * 32 + fn * 16 + lm;
                if (col < ldc) {
                    float v = acc[fm][fn].e[r];
                    if (HAS_BIAS) v += (col < nbias) ? bias[col] : 0.f;
                    if (RELU)     v = fmaxf(v, 0.f);
                    size_t o = (size_t)row * ldc + col;
                    if (OUT_BF16) ((bf16_t*)Cptr)[o] = (bf16_t)v;
                    else          ((float*)Cptr)[o]  = v;
                }
            }
}

// ---------------- batch norm ----------------
__global__ void k_bn_partial(const float* __restrict__ h2, float* __restrict__ sumbuf) {
    int c = threadIdx.x;                       // 0..319
    int chunk = (N_NODES + gridDim.x - 1) / gridDim.x;
    int i0 = blockIdx.x * chunk;
    int i1 = i0 + chunk; if (i1 > N_NODES) i1 = N_NODES;
    float s = 0.f, q = 0.f;
    for (int i = i0; i < i1; ++i) {
        float v = h2[(size_t)i * EMBP + c];
        s += v; q += v * v;
    }
    atomicAdd(&sumbuf[c], s);
    atomicAdd(&sumbuf[EMBP + c], q);
}

__global__ void k_bn_apply(const float* __restrict__ h2, const float* __restrict__ sumbuf,
                           const float* __restrict__ gamma, const float* __restrict__ beta,
                           float* __restrict__ h, int relu) {
    int idx = blockIdx.x * blockDim.x + threadIdx.x;
    if (idx >= N_NODES * EMB) return;
    int i = idx / EMB, c = idx % EMB;
    float m   = sumbuf[c] * (1.f / N_NODES);
    float var = sumbuf[EMBP + c] * (1.f / N_NODES) - m * m;
    float v = (h2[(size_t)i * EMBP + c] - m) * rsqrtf(var + BN_EPS) * gamma[c] + beta[c];
    if (relu) v = fmaxf(v, 0.f);
    h[(size_t)i * EMBP + c] = v;
}

// ---------------- vector quantization ----------------
__global__ void k_cbnorm(const float* __restrict__ cb, float* __restrict__ cbn,
                         float* __restrict__ lossacc) {
    int j = blockIdx.x * blockDim.x + threadIdx.x;
    if (j == 0) *lossacc = 0.f;                // zeroed before k_vq (stream order)
    if (j >= NUM_TOKENS) return;
    float s = 0.f;
    for (int c = 0; c < EMB; ++c) { float v = cb[j * EMB + c]; s += v * v; }
    cbn[j] = s;
}

// wave per node: masked argmin of cbn[j] - 2*S[i][j], then q = cb[idx]
__global__ __launch_bounds__(256)
void k_vq(const float* __restrict__ h, const float* __restrict__ S,
          const float* __restrict__ cbn, const float* __restrict__ cb,
          const int* __restrict__ x, float* __restrict__ out,
          float* __restrict__ lossacc) {
    int i = blockIdx.x * 8 + (threadIdx.x >> 5);
    if (i >= N_NODES) return;
    int lane = threadIdx.x & 31;
    int at = x[2 * i];
    int lo, hi;
    if      (at == 5) { lo = 0;   hi = 377; }
    else if (at == 6) { lo = 378; hi = 433; }
    else if (at == 7) { lo = 434; hi = 488; }
    else              { lo = 489; hi = 511; }
    const float* Si = S + (size_t)i * NUM_TOKENS;
    float best = 3.4e38f; int bidx = lo;
    for (int j = lo + lane; j < hi; j += 32) {
        float v = cbn[j] - 2.f * Si[j];
        if (v < best) { best = v; bidx = j; }
    }
    for (int off = 16; off > 0; off >>= 1) {
        float ob = __shfl_down(best, off);
        int   oi = __shfl_down(bidx, off);
        if (ob < best || (ob == best && oi < bidx)) { best = ob; bidx = oi; }
    }
    bidx = __shfl(bidx, 0);
    const float* q  = cb + (size_t)bidx * EMB;
    const float* ei = h + (size_t)i * EMBP;
    float ls = 0.f;
    for (int c = lane; c < EMB; c += 32) {
        float qc = q[c];
        out[(size_t)i * EMB + c] = qc;        // quantized = e + sg(q - e) == q in value
        float d = qc - ei[c];
        ls += d * d;
    }
    for (int off = 16; off > 0; off >>= 1) ls += __shfl_down(ls, off);
    if (lane == 0) atomicAdd(lossacc, ls);
}

__global__ void k_finalize(const float* __restrict__ lossacc, float* __restrict__ out) {
    if (blockIdx.x == 0 && threadIdx.x == 0)
        out[(size_t)N_NODES * EMB] =
            1.25f * lossacc[0] / (float)((size_t)N_NODES * EMB);  // (1+COMMIT)*mean
}

// ---------------- launcher ----------------
extern "C" void kernel_launch(void* const* d_in, const int* in_sizes, int n_in,
                              void* d_out, int out_size, void* d_ws, size_t ws_size,
                              hipStream_t stream) {
    (void)in_sizes; (void)n_in; (void)out_size; (void)ws_size;

    // leaves in setup_inputs() dict-insertion order
    const float* x_emb1 = (const float*)d_in[0];
    const float* x_emb2 = (const float*)d_in[1];
    struct Layer { const float *ee1, *ee2, *W1, *b1, *W2, *b2, *gamma, *beta; } L[NUM_LAYERS];
    for (int l = 0; l < NUM_LAYERS; ++l) {
        const int b = 2 + l * 8;
        L[l].ee1   = (const float*)d_in[b + 0];
        L[l].ee2   = (const float*)d_in[b + 1];
        L[l].W1    = (const float*)d_in[b + 2];
        L[l].b1    = (const float*)d_in[b + 3];
        L[l].W2    = (const float*)d_in[b + 4];
        L[l].b2    = (const float*)d_in[b + 5];
        L[l].gamma = (const float*)d_in[b + 6];
        L[l].beta  = (const float*)d_in[b + 7];
    }
    const float* codebook = (const float*)d_in[42];
    const int*   x        = (const int*)d_in[43];
    const int*   ei       = (const int*)d_in[44];
    const int*   ea       = (const int*)d_in[45];

    // workspace carve-up
    char* ws = (char*)d_ws;
    size_t off = 0;
    auto carve = [&](size_t bytes) -> void* {
        void* p = ws + off;
        off = (off + bytes + 255) & ~(size_t)255;
        return p;
    };
    float*  h    = (float*)carve((size_t)MP * EMBP * 4);
    float*  aggr = (float*)carve((size_t)MP * EMBP * 4);        // also serves as h2
    void*   big  = carve((size_t)MP * NUM_TOKENS * 4);          // Y(bf16,MPxYW) / S(f32,MPx512)
    bf16_t* Y    = (bf16_t*)big;
    float*  S    = (float*)big;
    bf16_t* W1t  = (bf16_t*)carve((size_t)NT1 * EMBP * 2);      // B^T panels
    bf16_t* W2t  = (bf16_t*)carve((size_t)NT2 * HIDP * 2);
    bf16_t* CBt  = (bf16_t*)carve((size_t)NUM_TOKENS * EMBP * 2);
    float*  sumbuf  = (float*)carve(2 * EMBP * 4);
    float*  cbn     = (float*)carve(NUM_TOKENS * 4);
    float*  lossacc = (float*)carve(4);

    const int TPB = 256;
    const int nHP = MP * EMBP;

    k_init_h<<<(nHP + TPB - 1) / TPB, TPB, 0, stream>>>(x, x_emb1, x_emb2, h);

    for (int l = 0; l < NUM_LAYERS; ++l) {
        k_aggr_init<<<(nHP + TPB - 1) / TPB, TPB, 0, stream>>>(h, L[l].ee1, L[l].ee2, aggr);
        k_scatter<<<(N_EDGES + 7) / 8, 256, 0, stream>>>(h, ei, ea, L[l].ee1, L[l].ee2, aggr);

        // GEMM1: Y = relu(aggr @ W1 + b1)   (A f32, out bf16, ldc=YW -> unguarded)
        k_pack_bt<<<(NT1 * EMBP + TPB - 1) / TPB, TPB, 0, stream>>>(
            L[l].W1, W1t, EMB, HID, EMBP, NT1, 1);
        dim3 g1(MP / 64, NT1 / 128);
        k_gemm<false, true, true, true><<<g1, 256, 0, stream>>>(
            aggr, EMBP, W1t, L[l].b1, HID, Y, YW, EMBP);

        // GEMM2: h2 = Y @ W2 + b2           (A bf16, out f32 into aggr)
        k_pack_bt<<<(NT2 * HIDP + TPB - 1) / TPB, TPB, 0, stream>>>(
            L[l].W2, W2t, HID, EMB, HIDP, NT2, 1);
        dim3 g2(MP / 64, NT2 / 128);
        k_gemm<true, true, false, false><<<g2, 256, 0, stream>>>(
            Y, YW, W2t, L[l].b2, EMB, aggr, EMBP, HIDP);

        k_zero_buf<<<(2 * EMBP + TPB - 1) / TPB, TPB, 0, stream>>>(sumbuf, 2 * EMBP);
        k_bn_partial<<<128, EMBP, 0, stream>>>(aggr, sumbuf);
        k_bn_apply<<<(N_NODES * EMB + TPB - 1) / TPB, TPB, 0, stream>>>(
            aggr, sumbuf, L[l].gamma, L[l].beta, h, l < NUM_LAYERS - 1 ? 1 : 0);
    }

    // S = e @ cb^T  (B^T == codebook itself, just pad to bf16)
    k_pack_bt<<<(NUM_TOKENS * EMBP + TPB - 1) / TPB, TPB, 0, stream>>>(
        codebook, CBt, EMB, NUM_TOKENS, EMBP, NUM_TOKENS, 0);
    dim3 g3(MP / 64, NUM_TOKENS / 128);
    k_gemm<false, false, false, false><<<g3, 256, 0, stream>>>(
        h, EMBP, CBt, nullptr, 0, S, NUM_TOKENS, EMBP);

    k_cbnorm<<<(NUM_TOKENS + TPB - 1) / TPB, TPB, 0, stream>>>(codebook, cbn, lossacc);
    k_vq<<<(N_NODES + 7) / 8, 256, 0, stream>>>(h, S, cbn, codebook, x, (float*)d_out, lossacc);
    k_finalize<<<1, 1, 0, stream>>>(lossacc, (float*)d_out);
}
